// EfficientGATConv_50276887167084
// MI455X (gfx1250) — matrix operand order repaired and checked by reading, not compiled, not run on hardware
//
#include <hip/hip_runtime.h>
#include <hip/hip_bf16.h>
#include <math.h>

typedef __attribute__((ext_vector_type(16))) __bf16 v16bf;
typedef __attribute__((ext_vector_type(8)))  float  v8f;

#define HD        128   // H*D = IN_FEATS
#define NEG_SLOPE 0.01f

// ---------------------------------------------------------------------------
// Kernel 1: h = feat @ W   (50000x128 * 128x128), bf16 WMMA, f32 accumulate.
// Block = 256 threads (8 waves). Each block computes a 128x128 tile of h;
// each wave owns a 16x128 strip = 8 C-tiles, K=128 in 4 WMMA steps of K=32.
// W is staged transposed in LDS so B fragments are contiguous 16B loads.
// ---------------------------------------------------------------------------
__global__ __launch_bounds__(256)
void gat_gemm_wmma(const float* __restrict__ feat, const float* __restrict__ W,
                   float* __restrict__ h, int N) {
  __shared__ __bf16 sA[128 * 128];   // feat tile, row-major [row][k]
  __shared__ __bf16 sWt[128 * 128];  // W transposed     [n][k]

  const int tid  = threadIdx.x;
  const int row0 = blockIdx.x * 128;

  // Stage W transposed (coalesced global reads, f32 -> bf16)
  #pragma unroll 4
  for (int i = 0; i < 64; ++i) {
    int idx = i * 256 + tid;          // idx = k*128 + n
    int k = idx >> 7, n = idx & 127;
    sWt[n * 128 + k] = (__bf16)W[idx];
  }
  // Stage feat tile (coalesced, zero-pad past N)
  #pragma unroll 4
  for (int i = 0; i < 64; ++i) {
    int idx = i * 256 + tid;          // idx = r*128 + c
    int r = idx >> 7, c = idx & 127;
    int node = row0 + r;
    float v = (node < N) ? feat[(size_t)node * HD + c] : 0.0f;
    sA[idx] = (__bf16)v;
  }
  __syncthreads();

  const int wave = tid >> 5;
  const int lane = tid & 31;
  const int hi   = lane >> 4;   // lane group 0/1
  const int l15  = lane & 15;

  v8f acc[8] = {};                 // 8 tiles of 16x16 f32

  union Frag { uint4 q[2]; v16bf v; };

  const int arow = wave * 16 + l15;      // A: M = lane&15 in both groups
  #pragma unroll
  for (int kt = 0; kt < 4; ++kt) {
    // A fragment (16x32 bf16): lanes 0-15 K=kt*32+{0..7,16..23},
    //                          lanes 16-31 K=kt*32+{8..15,24..31}
    Frag A;
    const int kbA = kt * 32 + hi * 8;
    A.q[0] = *(const uint4*)(&sA[arow * 128 + kbA]);
    A.q[1] = *(const uint4*)(&sA[arow * 128 + kbA + 16]);
    #pragma unroll
    for (int nt = 0; nt < 8; ++nt) {
      // B fragment (32x16 bf16): lanes 0-15 K=kt*32+{0..7,16..23}, lanes 16-31 K=+{8..15,24..31}
      Frag B;
      const int kbB = kt * 32 + hi * 8;
      B.q[0] = *(const uint4*)(&sWt[(nt * 16 + l15) * 128 + kbB]);
      B.q[1] = *(const uint4*)(&sWt[(nt * 16 + l15) * 128 + kbB + 16]);
      acc[nt] = __builtin_amdgcn_wmma_f32_16x16x32_bf16(
          false, A.v, false, B.v, (short)0, acc[nt], false, false);
    }
  }

  // C layout: VGPR v -> M=v (lanes 0-15) / M=v+8 (lanes 16-31), N=lane&15
  #pragma unroll
  for (int nt = 0; nt < 8; ++nt) {
    #pragma unroll
    for (int v = 0; v < 8; ++v) {
      int row = row0 + wave * 16 + v + hi * 8;
      int col = nt * 16 + l15;
      if (row < N) h[(size_t)row * HD + col] = acc[nt][v];
    }
  }
}

// ---------------------------------------------------------------------------
// Kernel 2: el[n,h] = h[n,h,:].attn_l[h]   er[n,h] = h[n,h,:].attn_r[h]
// ---------------------------------------------------------------------------
__global__ void gat_el_er(const float* __restrict__ h,
                          const float* __restrict__ attn_l,
                          const float* __restrict__ attn_r,
                          float* __restrict__ el, float* __restrict__ er, int N) {
  int t = blockIdx.x * blockDim.x + threadIdx.x;
  if (t >= N * 4) return;
  int n = t >> 2, hd = t & 3;
  const float4* hp = (const float4*)(h + (size_t)n * HD + hd * 32);
  const float4* al = (const float4*)(attn_l + hd * 32);
  const float4* ar = (const float4*)(attn_r + hd * 32);
  float sl = 0.f, sr = 0.f;
  #pragma unroll
  for (int i = 0; i < 8; ++i) {
    float4 hv = hp[i], a = al[i], b = ar[i];
    sl += hv.x * a.x + hv.y * a.y + hv.z * a.z + hv.w * a.w;
    sr += hv.x * b.x + hv.y * b.y + hv.z * b.z + hv.w * b.w;
  }
  el[t] = sl;
  er[t] = sr;
}

// ---------------------------------------------------------------------------
// Kernel 3: out = feat  (residual init; atomics accumulate on top)
// ---------------------------------------------------------------------------
__global__ void gat_init_out(const float* __restrict__ feat,
                             float* __restrict__ out, int total) {
  int t = blockIdx.x * blockDim.x + threadIdx.x;
  if (t < total) out[t] = feat[t];
}

// ---------------------------------------------------------------------------
// Kernel 4: per-edge leaky_relu + softmax over the 4 heads -> attn[E,4]
// ---------------------------------------------------------------------------
__global__ void gat_edge_softmax(const float* __restrict__ el,
                                 const float* __restrict__ er,
                                 const int* __restrict__ src,
                                 const int* __restrict__ dst,
                                 float* __restrict__ attn, int E) {
  int e = blockIdx.x * blockDim.x + threadIdx.x;
  if (e >= E) return;
  int s = src[e], d = dst[e];
  float4 lv = *(const float4*)(el + (size_t)s * 4);
  float4 rv = *(const float4*)(er + (size_t)d * 4);
  float l0 = lv.x + rv.x, l1 = lv.y + rv.y, l2 = lv.z + rv.z, l3 = lv.w + rv.w;
  l0 = l0 > 0.f ? l0 : NEG_SLOPE * l0;
  l1 = l1 > 0.f ? l1 : NEG_SLOPE * l1;
  l2 = l2 > 0.f ? l2 : NEG_SLOPE * l2;
  l3 = l3 > 0.f ? l3 : NEG_SLOPE * l3;
  float m = fmaxf(fmaxf(l0, l1), fmaxf(l2, l3));
  float x0 = __expf(l0 - m), x1 = __expf(l1 - m);
  float x2 = __expf(l2 - m), x3 = __expf(l3 - m);
  float inv = 1.0f / (x0 + x1 + x2 + x3);
  float4 a = make_float4(x0 * inv, x1 * inv, x2 * inv, x3 * inv);
  *(float4*)(attn + (size_t)e * 4) = a;
}

// ---------------------------------------------------------------------------
// Kernel 5: thread per (edge, feature): gather h[src], scale by attn,
// atomic-add into out[dst]. h/out/attn are all L2-resident (192 MB L2).
// ---------------------------------------------------------------------------
__global__ __launch_bounds__(256)
void gat_edge_scatter(const float* __restrict__ h,
                      const float* __restrict__ attn,
                      const int* __restrict__ src,
                      const int* __restrict__ dst,
                      float* __restrict__ out, int E) {
  long long t = (long long)blockIdx.x * blockDim.x + threadIdx.x;
  if (t >= (long long)E * HD) return;
  int e  = (int)(t >> 7);
  int f  = (int)(t & (HD - 1));
  int hd = f >> 5;
  int s = src[e], d = dst[e];
  float a   = attn[(size_t)e * 4 + hd];
  float val = h[(size_t)s * HD + f] * a;
  atomicAdd(&out[(size_t)d * HD + f], val);
}

// ---------------------------------------------------------------------------
// Kernel 6: in-place ELU
// ---------------------------------------------------------------------------
__global__ void gat_elu(float* __restrict__ out, int total) {
  int t = blockIdx.x * blockDim.x + threadIdx.x;
  if (t >= total) return;
  float x = out[t];
  out[t] = x > 0.f ? x : (__expf(x) - 1.0f);
}

extern "C" void kernel_launch(void* const* d_in, const int* in_sizes, int n_in,
                              void* d_out, int out_size, void* d_ws, size_t ws_size,
                              hipStream_t stream) {
  const float* feat   = (const float*)d_in[0];
  const int*   src    = (const int*)d_in[1];
  const int*   dst    = (const int*)d_in[2];
  const float* W_fc   = (const float*)d_in[3];
  const float* attn_l = (const float*)d_in[4];
  const float* attn_r = (const float*)d_in[5];

  const int N = in_sizes[0] / HD;
  const int E = in_sizes[1];

  float* out  = (float*)d_out;
  float* h    = (float*)d_ws;                 // N*128 f32
  float* el   = h  + (size_t)N * HD;          // N*4
  float* er   = el + (size_t)N * 4;           // N*4
  float* attn = er + (size_t)N * 4;           // E*4

  // 1) projection GEMM (WMMA bf16 -> f32)
  gat_gemm_wmma<<<(N + 127) / 128, 256, 0, stream>>>(feat, W_fc, h, N);
  // 2) per-node attention logits
  gat_el_er<<<(N * 4 + 255) / 256, 256, 0, stream>>>(h, attn_l, attn_r, el, er, N);
  // 3) out = feat (residual base)
  gat_init_out<<<(N * HD + 255) / 256, 256, 0, stream>>>(feat, out, N * HD);
  // 4) per-edge softmax over heads
  gat_edge_softmax<<<(E + 255) / 256, 256, 0, stream>>>(el, er, src, dst, attn, E);
  // 5) gather/scale/scatter-add
  long long tot = (long long)E * HD;
  gat_edge_scatter<<<(int)((tot + 255) / 256), 256, 0, stream>>>(h, attn, src, dst, out, E);
  // 6) ELU in place
  gat_elu<<<(N * HD + 255) / 256, 256, 0, stream>>>(out, N * HD);
}
